// TPGRUCell_17841294147974
// MI455X (gfx1250) — matrix operand order repaired and checked by reading, not compiled, not run on hardware
//
#include <hip/hip_runtime.h>
#include <hip/hip_bf16.h>
#include <math.h>
#include <stdint.h>

// ---------------- problem dims ----------------
#define T_LEN 512
#define BATCH 64
#define DIN   512
#define DH    512
#define DT    128
#define M_ROWS (T_LEN * BATCH)   // 32768

typedef __bf16 bf16_t;
typedef __attribute__((ext_vector_type(16))) __bf16 bf16x16;
typedef __attribute__((ext_vector_type(8)))  float  f32x8;
typedef __attribute__((ext_vector_type(4)))  int    v4i_t;

#define AS1 __attribute__((address_space(1)))
#define AS3 __attribute__((address_space(3)))

#if __has_builtin(__builtin_amdgcn_global_load_async_to_lds_b128) && \
    __has_builtin(__builtin_amdgcn_s_wait_asynccnt)
#define HAVE_ASYNC_LDS 1
#endif
#if __has_builtin(__builtin_amdgcn_tensor_load_to_lds) && \
    __has_builtin(__builtin_amdgcn_s_wait_tensorcnt)
#define HAVE_TDM 1
#endif

// ---------------- workspace layout (bytes) ----------------
static constexpr size_t OFF_XBF   = 0;                          // 32768*512 bf16
static constexpr size_t OFF_WT_P1 = OFF_XBF   + 33554432;       // [1536][512] bf16
static constexpr size_t OFF_WT_P2 = OFF_WT_P1 + 1572864;
static constexpr size_t OFF_WT_G1 = OFF_WT_P2 + 1572864;
static constexpr size_t OFF_WT_G2 = OFF_WT_G1 + 1572864;
static constexpr size_t OFF_TPA   = OFF_WT_G2 + 1572864;        // [64][1536] f32
static constexpr size_t OFF_TPB   = OFF_TPA   + 393216;
static constexpr size_t OFF_ZRXS  = OFF_TPB   + 393216;         // 32768*1024 bf16
static constexpr size_t OFF_HXS   = OFF_ZRXS  + 67108864;       // 32768*512 bf16
static constexpr size_t OFF_ZRXTP = OFF_HXS   + 33554432;       // 32768*1024 bf16
static constexpr size_t OFF_HXTP  = OFF_ZRXTP + 67108864;       // 32768*512 bf16
static constexpr size_t OFF_HF32  = OFF_HXTP  + 33554432;       // 64*512 f32
static constexpr size_t OFF_HBF   = OFF_HF32  + 131072;         // 64*512 bf16
static constexpr size_t OFF_TMPZR = OFF_HBF   + 65536;          // 64*1024 bf16
static constexpr size_t OFF_TMPHH = OFF_TMPZR + 131072;         // 64*512 bf16
static constexpr size_t OFF_BAR   = OFF_TMPHH + 65536;          // barrier counter

static constexpr unsigned RECUR_WGS = 16;
static constexpr size_t   LDS_BYTES = (96 * 512 + 3 * 32 * 512) * sizeof(bf16_t); // 192 KB

// ---------------- device helpers ----------------
__device__ __forceinline__ bf16x16 load_frag(const bf16_t* p0, const bf16_t* p1) {
  union { bf16x16 v; uint4 q[2]; } u;
  u.q[0] = *reinterpret_cast<const uint4*>(p0);
  u.q[1] = *reinterpret_cast<const uint4*>(p1);
  return u.v;
}

__device__ __forceinline__ f32x8 wmma_bf(bf16x16 a, bf16x16 b, f32x8 c) {
  return __builtin_amdgcn_wmma_f32_16x16x32_bf16(false, a, false, b, (short)0, c, false, false);
}

__device__ __forceinline__ float sigmoidf_dev(float x) { return 1.0f / (1.0f + __expf(-x)); }

__device__ __forceinline__ void grid_sync(unsigned* bar, unsigned goal) {
  __threadfence();
  __syncthreads();
  if (threadIdx.x == 0) {
    atomicAdd(bar, 1u);
    while (*(volatile unsigned*)bar < goal) { __builtin_amdgcn_s_sleep(4); }
  }
  __syncthreads();
  __threadfence();
}

// 16B global -> LDS, async (ASYNCcnt) when available
__device__ __forceinline__ void async_copy_b128(const void* g, void* l) {
#if defined(HAVE_ASYNC_LDS)
  __builtin_amdgcn_global_load_async_to_lds_b128(
      (AS1 v4i_t*)(uintptr_t)g,
      (AS3 v4i_t*)(uint32_t)(uintptr_t)l, 0, 0);
#else
  *reinterpret_cast<uint4*>(l) = *reinterpret_cast<const uint4*>(g);
#endif
}
__device__ __forceinline__ void async_wait_all() {
#if defined(HAVE_ASYNC_LDS)
  __builtin_amdgcn_s_wait_asynccnt(0);
#endif
}

#if defined(HAVE_TDM)
typedef __attribute__((ext_vector_type(4))) unsigned int u32x4;
typedef __attribute__((ext_vector_type(8))) int i32x8;
typedef __attribute__((ext_vector_type(4))) int i32x4;

// Linear DMA: nbytes (multiple of 8, tile_dim0 <= 65535 elems) global -> LDS via the
// Tensor Data Mover.  D# per CDNA5 ISA:
//   group0: count=1 | lds_addr | global_addr[56:0] | type=2
//   group1: data_size=8B; 1-row tile: tensor_dim0=tile_dim0=elems, tensor_dim1=tile_dim1=1
__device__ __forceinline__ void tdm_load_linear(const void* gsrc, void* ldst, unsigned nbytes) {
  const unsigned elems = nbytes >> 3;                       // 8-byte elements
  const unsigned long long ga = (unsigned long long)(uintptr_t)gsrc;
  const unsigned lofs = (unsigned)(uintptr_t)ldst;          // LDS byte offset
  u32x4 g0 = { 1u,                                          // count=1 (valid, user mode)
               lofs,                                        // lds_addr
               (unsigned)ga,                                // global_addr[31:0]
               (unsigned)((ga >> 32) & 0x01ffffffu) | (2u << 30) };  // addr[56:32] | type=2
  i32x8 g1 = { (int)(3u << 16),                             // wg_mask=0, data_size=3 (8B)
               (int)((elems & 0xffffu) << 16),              // tensor_dim0[15:0]
               (int)(((elems >> 16) & 0xffffu) | (1u << 16)), // tensor_dim0[31:16] | tensor_dim1=1
               (int)((elems & 0xffffu) << 16),              // tile_dim0 = elems
               1,                                           // tile_dim1 = 1 (tile_dim2 unused)
               (int)elems,                                  // tensor_dim0_stride[31:0]
               0, 0 };
  i32x4 gz = { 0, 0, 0, 0 };
#if __clang_major__ >= 23
  i32x8 gz8 = { 0, 0, 0, 0, 0, 0, 0, 0 };
  __builtin_amdgcn_tensor_load_to_lds(g0, g1, gz, gz, gz8, 0);
#else
  __builtin_amdgcn_tensor_load_to_lds(g0, g1, gz, gz, 0);
#endif
}
#endif

// ---------------- precompute kernels ----------------
__global__ void k_convert_x(const float* __restrict__ src, bf16_t* __restrict__ dst, int n) {
  for (int i = blockIdx.x * blockDim.x + threadIdx.x; i < n; i += gridDim.x * blockDim.x)
    dst[i] = (bf16_t)src[i];
}

// dst[n*K + k] = (bf16) src[k*N + n]   (transpose to N-major for the WMMA B layout)
__global__ void k_conv_transpose(const float* __restrict__ src, bf16_t* __restrict__ dst,
                                 int K, int N) {
  int idx = blockIdx.x * blockDim.x + threadIdx.x;
  if (idx >= K * N) return;
  int n = idx / K, k = idx % K;
  dst[idx] = (bf16_t)src[(size_t)k * N + n];
}

// topic projections: tpA = [topic@W_tp2zr_x | topic@W_tp2h_x], tpB = [..zr_h | ..h_h]
__global__ void k_topic(const float* __restrict__ topic,
                        const float* __restrict__ Wax, const float* __restrict__ Wah,
                        const float* __restrict__ Wbx, const float* __restrict__ Wbh,
                        float* __restrict__ tpA, float* __restrict__ tpB) {
  int idx = blockIdx.x * blockDim.x + threadIdx.x;
  if (idx >= BATCH * 1536) return;
  int b = idx / 1536, j = idx % 1536;
  const float* trow = topic + (size_t)b * DT;
  float sa = 0.f, sb = 0.f;
  if (j < 1024) {
    for (int k = 0; k < DT; ++k) { float t = trow[k]; sa += t * Wax[(size_t)k * 1024 + j]; sb += t * Wbx[(size_t)k * 1024 + j]; }
  } else {
    int jj = j - 1024;
    for (int k = 0; k < DT; ++k) { float t = trow[k]; sa += t * Wah[(size_t)k * 512 + jj]; sb += t * Wbh[(size_t)k * 512 + jj]; }
  }
  tpA[idx] = sa; tpB[idx] = sb;
}

// P1: [ (x@W_x2zr)*tp_zr_x | (x@W_x2h)*tp_h_x ]  -> bf16  (M=32768, K=512, Nfused=1536)
__global__ void __launch_bounds__(256)
k_gemm_p1(const bf16_t* __restrict__ xbf, const bf16_t* __restrict__ Wt,
          const float* __restrict__ tpA, bf16_t* __restrict__ zrxs, bf16_t* __restrict__ hxs) {
  const int lane = threadIdx.x & 31;
  const int lo = lane & 15, kh = lane >> 4;
  const int wave = (blockIdx.x * blockDim.x + threadIdx.x) >> 5;
  const int nwaves = (gridDim.x * blockDim.x) >> 5;
  const int NT = 1536 / 16;
  const int total = (M_ROWS / 16) * NT;
  for (int tt = wave; tt < total; tt += nwaves) {
    int mt = tt / NT, nt = tt % NT;
    int m0 = mt * 16, n0 = nt * 16;
    f32x8 acc = {};
    const bf16_t* Ab = xbf + (size_t)(m0 + lo) * DIN;
    const bf16_t* Bb = Wt + (size_t)(n0 + lo) * DIN;
    #pragma unroll 4
    for (int k0 = 0; k0 < DIN; k0 += 32) {
      __builtin_prefetch(Ab + k0 + 128, 0, 1);
      bf16x16 a = load_frag(Ab + k0 + 8 * kh, Ab + k0 + 16 + 8 * kh);
      bf16x16 b = load_frag(Bb + k0 + 8 * kh, Bb + k0 + 16 + 8 * kh);
      acc = wmma_bf(a, b, acc);
    }
    const int radd = (lane >= 16) ? 8 : 0;
    const int n = n0 + lo;
    // n0 is 16-aligned and the 1024 split is tile-aligned -> wave-uniform branch
    bf16_t* dbase; size_t dstride;
    if (n0 < 1024) { dbase = zrxs + n;          dstride = 1024; }
    else           { dbase = hxs + (n - 1024);  dstride = 512;  }
    const float* tpcol = tpA + n;
    #pragma unroll
    for (int e = 0; e < 8; ++e) {
      int m = m0 + e + radd;
      float v = acc[e] * tpcol[(size_t)(m & 63) * 1536];
      dbase[(size_t)m * dstride] = (bf16_t)v;
    }
  }
}

// P2: [ Zs@W_xtp2z+bz | Rs@W_xtp2r+br | Hx@W_xtp2h+bh ] -> bf16 (M=32768, K=512, Nfused=1536)
__global__ void __launch_bounds__(256)
k_gemm_p2(const bf16_t* __restrict__ zrxs, const bf16_t* __restrict__ hxs,
          const bf16_t* __restrict__ Wt,
          const float* __restrict__ bz, const float* __restrict__ br, const float* __restrict__ bh,
          bf16_t* __restrict__ zrxtp, bf16_t* __restrict__ hxtp) {
  const int lane = threadIdx.x & 31;
  const int lo = lane & 15, kh = lane >> 4;
  const int wave = (blockIdx.x * blockDim.x + threadIdx.x) >> 5;
  const int nwaves = (gridDim.x * blockDim.x) >> 5;
  const int NT = 1536 / 16;
  const int total = (M_ROWS / 16) * NT;
  for (int tt = wave; tt < total; tt += nwaves) {
    int mt = tt / NT, nt = tt % NT;
    int m0 = mt * 16, n0 = nt * 16;
    const bf16_t* Ab;
    if (n0 < 512)       Ab = zrxs + (size_t)(m0 + lo) * 1024;
    else if (n0 < 1024) Ab = zrxs + (size_t)(m0 + lo) * 1024 + 512;
    else                Ab = hxs + (size_t)(m0 + lo) * 512;
    const bf16_t* Bb = Wt + (size_t)(n0 + lo) * 512;
    f32x8 acc = {};
    #pragma unroll 4
    for (int k0 = 0; k0 < 512; k0 += 32) {
      bf16x16 a = load_frag(Ab + k0 + 8 * kh, Ab + k0 + 16 + 8 * kh);
      bf16x16 b = load_frag(Bb + k0 + 8 * kh, Bb + k0 + 16 + 8 * kh);
      acc = wmma_bf(a, b, acc);
    }
    const int radd = (lane >= 16) ? 8 : 0;
    const int n = n0 + lo;
    float bias = (n < 512) ? bz[n] : (n < 1024) ? br[n - 512] : bh[n - 1024];
    bf16_t* dbase; size_t dstride;
    if (n0 < 1024) { dbase = zrxtp + n;          dstride = 1024; }
    else           { dbase = hxtp + (n - 1024);  dstride = 512;  }
    #pragma unroll
    for (int e = 0; e < 8; ++e) {
      int m = m0 + e + radd;
      dbase[(size_t)m * dstride] = (bf16_t)(acc[e] + bias);
    }
  }
}

// ---------------- persistent recurrence kernel ----------------
// 16 WGs; WG w owns output columns [w*96, w*96+96) of the fused G1 space (1536) and
// [w*32, w*32+32) of the G2/h space (512). All recurrent weight slices live in LDS (192KB).
__global__ void __launch_bounds__(256)
k_recur(const bf16_t* __restrict__ WtG1, const bf16_t* __restrict__ WtG2,
        const float* __restrict__ tpB,
        const bf16_t* __restrict__ zrxtp, const bf16_t* __restrict__ hxtp,
        const float* __restrict__ mask,
        float* __restrict__ h_f32, bf16_t* __restrict__ h_bf,
        bf16_t* __restrict__ tmp_zr, bf16_t* __restrict__ tmp_hh,
        unsigned* __restrict__ bar, float* __restrict__ out, float* __restrict__ outHT) {
  extern __shared__ bf16_t lds[];
  const int w = blockIdx.x;
  const int tid = threadIdx.x;
  const int lane = tid & 31;
  const int waveId = tid >> 5;
  const int lo = lane & 15, kh = lane >> 4;

  // --- preload weight slices into LDS ---
  // G1 slice (96KB, contiguous): one TDM descriptor per WGP when available.
#if defined(HAVE_TDM)
  if (waveId == 0) {
    tdm_load_linear(WtG1 + (size_t)w * 96 * 512, lds, 96 * 512 * sizeof(bf16_t));
  }
#else
  {
    const uint4* s1 = reinterpret_cast<const uint4*>(WtG1 + (size_t)w * 96 * 512);
    uint4* d1 = reinterpret_cast<uint4*>(lds);
    for (int i = tid; i < 96 * 512 / 8; i += 256) async_copy_b128(s1 + i, d1 + i);
  }
#endif
  // G2 slices (3 x 32KB): async global->LDS (no VGPR bounce).
  for (int s = 0; s < 3; ++s) {
    const uint4* s2 = reinterpret_cast<const uint4*>(WtG2 + ((size_t)s * 512 + w * 32) * 512);
    uint4* d2 = reinterpret_cast<uint4*>(lds + 96 * 512 + s * 32 * 512);
    for (int i = tid; i < 32 * 512 / 8; i += 256) async_copy_b128(s2 + i, d2 + i);
  }
  async_wait_all();
#if defined(HAVE_TDM)
  __builtin_amdgcn_s_wait_tensorcnt(0);
#endif
  __syncthreads();
  __builtin_amdgcn_s_cluster_barrier();   // NOP unless dispatched as a cluster

  const bf16_t* ldsG1 = lds;
  const bf16_t* ldsG2 = lds + 96 * 512;
  unsigned epoch = 0;

  for (int t = 0; t < T_LEN; ++t) {
    // ---- G1: tmp_zr = (h@W_h2zr)*tp_zr_h ; tmp_hh = (h@W_h2h)*tp_h_h ----
    #pragma unroll
    for (int i = 0; i < 3; ++i) {
      int l = waveId * 3 + i;           // 0..23 : 4 M-tiles x 6 local N-tiles
      int mt = l / 6, ntl = l % 6;
      int m0 = mt * 16, nloc0 = ntl * 16;
      int n0g = w * 96 + nloc0;
      int n = n0g + lo;                 // fused column 0..1535
      f32x8 acc = {};
      const bf16_t* Ab = h_bf + (size_t)(m0 + lo) * DH;
      const bf16_t* Bb = ldsG1 + (size_t)(nloc0 + lo) * DH;
      for (int k0 = 0; k0 < DH; k0 += 32) {
        bf16x16 a = load_frag(Ab + k0 + 8 * kh, Ab + k0 + 16 + 8 * kh);
        bf16x16 b = load_frag(Bb + k0 + 8 * kh, Bb + k0 + 16 + 8 * kh);
        acc = wmma_bf(a, b, acc);
      }
      int radd = (lane >= 16) ? 8 : 0;
      bf16_t* dbase; size_t dstride;     // tile-aligned 1024 split -> wave-uniform
      if (n0g < 1024) { dbase = tmp_zr + n;          dstride = 1024; }
      else            { dbase = tmp_hh + (n - 1024); dstride = 512;  }
      #pragma unroll
      for (int e = 0; e < 8; ++e) {
        int b = m0 + e + radd;
        dbase[(size_t)b * dstride] = (bf16_t)(acc[e] * tpB[(size_t)b * 1536 + n]);
      }
    }
    grid_sync(bar, (++epoch) * RECUR_WGS);

    // ---- G2: z2=z_htp@W_htp2z, r2=r_htp@W_htp2r, hc=hh@W_htp2h + fused gate update ----
    {
      int mt = waveId >> 1, ntl = waveId & 1;
      int m0 = mt * 16, nloc0 = ntl * 16;
      int n = w * 32 + nloc0 + lo;      // output column 0..511
      f32x8 accZ = {}, accR = {}, accH = {};
      const bf16_t* Az = tmp_zr + (size_t)(m0 + lo) * 1024;
      const bf16_t* Ar = Az + 512;
      const bf16_t* Ah = tmp_hh + (size_t)(m0 + lo) * 512;
      const bf16_t* Bz = ldsG2 + (size_t)(0 * 32 * 512) + (size_t)(nloc0 + lo) * 512;
      const bf16_t* Br = ldsG2 + (size_t)(1 * 32 * 512) + (size_t)(nloc0 + lo) * 512;
      const bf16_t* Bh = ldsG2 + (size_t)(2 * 32 * 512) + (size_t)(nloc0 + lo) * 512;
      for (int k0 = 0; k0 < 512; k0 += 32) {
        int ka = k0 + 8 * kh, kb = k0 + 8 * kh;
        accZ = wmma_bf(load_frag(Az + ka, Az + ka + 16), load_frag(Bz + kb, Bz + kb + 16), accZ);
        accR = wmma_bf(load_frag(Ar + ka, Ar + ka + 16), load_frag(Br + kb, Br + kb + 16), accR);
        accH = wmma_bf(load_frag(Ah + ka, Ah + ka + 16), load_frag(Bh + kb, Bh + kb + 16), accH);
      }
      int radd = (lane >= 16) ? 8 : 0;
      #pragma unroll
      for (int e = 0; e < 8; ++e) {
        int b = m0 + e + radd;
        size_t row = (size_t)t * BATCH + b;
        float z = sigmoidf_dev((float)zrxtp[row * 1024 + n] + accZ[e]);
        float r = sigmoidf_dev((float)zrxtp[row * 1024 + 512 + n] + accR[e]);
        float hn = tanhf((float)hxtp[row * 512 + n] + r * accH[e]);
        float hold = h_f32[(size_t)b * 512 + n];
        float h = (1.f - z) * hn + z * hold;
        float mv = mask[row];
        h = mv * h + (1.f - mv) * hold;
        h_f32[(size_t)b * 512 + n] = h;
        h_bf[(size_t)b * 512 + n] = (bf16_t)h;
        out[row * 512 + n] = h;
        if (t == T_LEN - 1) outHT[(size_t)b * 512 + n] = h;
      }
    }
    grid_sync(bar, (++epoch) * RECUR_WGS);
  }
}

// ---------------- host launcher ----------------
extern "C" void kernel_launch(void* const* d_in, const int* in_sizes, int n_in,
                              void* d_out, int out_size, void* d_ws, size_t ws_size,
                              hipStream_t stream) {
  (void)in_sizes; (void)n_in; (void)out_size; (void)ws_size;
  const float* x         = (const float*)d_in[0];
  const float* topic     = (const float*)d_in[1];
  const float* mask      = (const float*)d_in[2];
  const float* W_x2zr    = (const float*)d_in[3];
  const float* W_tp2zr_x = (const float*)d_in[4];
  const float* W_xtp2z   = (const float*)d_in[5];
  const float* b_xtp2z   = (const float*)d_in[6];
  const float* W_xtp2r   = (const float*)d_in[7];
  const float* b_xtp2r   = (const float*)d_in[8];
  const float* W_h2zr    = (const float*)d_in[9];
  const float* W_tp2zr_h = (const float*)d_in[10];
  const float* W_htp2z   = (const float*)d_in[11];
  const float* W_htp2r   = (const float*)d_in[12];
  const float* W_x2h     = (const float*)d_in[13];
  const float* W_tp2h_x  = (const float*)d_in[14];
  const float* W_xtp2h   = (const float*)d_in[15];
  const float* b_xtp2h   = (const float*)d_in[16];
  const float* W_h2h     = (const float*)d_in[17];
  const float* W_tp2h_h  = (const float*)d_in[18];
  const float* W_htp2h   = (const float*)d_in[19];

  char* ws = (char*)d_ws;
  bf16_t* xbf   = (bf16_t*)(ws + OFF_XBF);
  bf16_t* WtP1  = (bf16_t*)(ws + OFF_WT_P1);
  bf16_t* WtP2  = (bf16_t*)(ws + OFF_WT_P2);
  bf16_t* WtG1  = (bf16_t*)(ws + OFF_WT_G1);
  bf16_t* WtG2  = (bf16_t*)(ws + OFF_WT_G2);
  float*  tpA   = (float*)(ws + OFF_TPA);
  float*  tpB   = (float*)(ws + OFF_TPB);
  bf16_t* zrxs  = (bf16_t*)(ws + OFF_ZRXS);
  bf16_t* hxs   = (bf16_t*)(ws + OFF_HXS);
  bf16_t* zrxtp = (bf16_t*)(ws + OFF_ZRXTP);
  bf16_t* hxtp  = (bf16_t*)(ws + OFF_HXTP);
  float*  hf32  = (float*)(ws + OFF_HF32);
  bf16_t* hbf   = (bf16_t*)(ws + OFF_HBF);
  bf16_t* tmpzr = (bf16_t*)(ws + OFF_TMPZR);
  bf16_t* tmphh = (bf16_t*)(ws + OFF_TMPHH);
  unsigned* bar = (unsigned*)(ws + OFF_BAR);

  float* out   = (float*)d_out;
  float* outHT = out + (size_t)T_LEN * BATCH * DH;

  (void)hipMemsetAsync(hf32, 0, BATCH * DH * sizeof(float), stream);
  (void)hipMemsetAsync(hbf, 0, BATCH * DH * sizeof(bf16_t), stream);
  (void)hipMemsetAsync(bar, 0, 256, stream);

  // input + weight conversion (f32 -> bf16, weights transposed to N-major)
  k_convert_x<<<4096, 256, 0, stream>>>(x, xbf, M_ROWS * DIN);
  auto convT = [&](const float* src, bf16_t* dst, int K, int N) {
    k_conv_transpose<<<(K * N + 255) / 256, 256, 0, stream>>>(src, dst, K, N);
  };
  convT(W_x2zr,  WtP1,              512, 1024);
  convT(W_x2h,   WtP1 + 1024 * 512, 512, 512);
  convT(W_xtp2z, WtP2,              512, 512);
  convT(W_xtp2r, WtP2 + 512 * 512,  512, 512);
  convT(W_xtp2h, WtP2 + 1024 * 512, 512, 512);
  convT(W_h2zr,  WtG1,              512, 1024);
  convT(W_h2h,   WtG1 + 1024 * 512, 512, 512);
  convT(W_htp2z, WtG2,              512, 512);
  convT(W_htp2r, WtG2 + 512 * 512,  512, 512);
  convT(W_htp2h, WtG2 + 1024 * 512, 512, 512);

  k_topic<<<(BATCH * 1536 + 255) / 256, 256, 0, stream>>>(
      topic, W_tp2zr_x, W_tp2h_x, W_tp2zr_h, W_tp2h_h, tpA, tpB);

  k_gemm_p1<<<1024, 256, 0, stream>>>(xbf, WtP1, tpA, zrxs, hxs);
  k_gemm_p2<<<1024, 256, 0, stream>>>(zrxs, hxs, WtP2, b_xtp2z, b_xtp2r, b_xtp2h, zrxtp, hxtp);

  (void)hipFuncSetAttribute(reinterpret_cast<const void*>(k_recur),
                            hipFuncAttributeMaxDynamicSharedMemorySize, (int)LDS_BYTES);
  k_recur<<<RECUR_WGS, 256, LDS_BYTES, stream>>>(
      WtG1, WtG2, tpB, zrxtp, hxtp, mask, hf32, hbf, tmpzr, tmphh, bar, out, outHT);
}
